// DynamicHypergraphTower_34952443855216
// MI455X (gfx1250) — matrix operand order repaired and checked by reading, not compiled, not run on hardware
//
#include <hip/hip_runtime.h>

// ---------------------------------------------------------------------------
// DynamicHypergraphTower for MI455X (gfx1250, wave32, WMMA bf16 + async LDS)
// ---------------------------------------------------------------------------
constexpr int CB = 16;    // batch
constexpr int CS = 64;    // scene steps
constexpr int CN = 100;   // entities
constexpr int CD = 1024;  // hidden

typedef __attribute__((ext_vector_type(16))) __bf16 v16bf;
typedef __attribute__((ext_vector_type(4)))  __bf16 v4bf;
typedef __attribute__((ext_vector_type(2)))  __bf16 v2bf;
typedef __attribute__((ext_vector_type(8)))  float  v8f;

__device__ __forceinline__ v8f wmma_bf16(const v16bf& a, const v16bf& b, const v8f& c) {
  return __builtin_amdgcn_wmma_f32_16x16x32_bf16(false, a, false, b, (short)0, c, false, false);
}

// CDNA5 async global->LDS copy (16B per lane), tracked by ASYNCcnt
__device__ __forceinline__ void async_load_lds_b128(unsigned lds_addr, const void* gaddr) {
  asm volatile("global_load_async_to_lds_b128 %0, %1, off"
               :: "v"(lds_addr), "v"(gaddr) : "memory");
}
__device__ __forceinline__ void wait_async0() {
  asm volatile("s_wait_asynccnt 0x0" ::: "memory");
}

// ---------------------------------------------------------------------------
// prep kernels
// ---------------------------------------------------------------------------
__global__ void k_stream(const float* __restrict__ sw, float* __restrict__ out3) {
  if (threadIdx.x == 0) {
    float a = sw[0], b = sw[1], c = sw[2];
    float mx = fmaxf(a, fmaxf(b, c));
    float ea = __expf(a - mx), eb = __expf(b - mx), ec = __expf(c - mx);
    float s = ea + eb + ec;
    out3[0] = ea / s; out3[1] = eb / s; out3[2] = ec / s;
  }
}

__global__ void k_convert(__bf16* __restrict__ dst, const float* __restrict__ src,
                          int n, const float* __restrict__ stream3, int sidx) {
  int i = blockIdx.x * blockDim.x + threadIdx.x;
  if (i < n) {
    float scale = (sidx >= 0) ? stream3[sidx] : 1.0f;
    dst[i] = (__bf16)(src[i] * scale);
  }
}

__global__ void k_bias2(float* __restrict__ dst, const float* __restrict__ b1,
                        const float* __restrict__ b2, const float* __restrict__ stream3, int n) {
  int i = blockIdx.x * blockDim.x + threadIdx.x;
  if (i < n) dst[i] = stream3[1] * b1[i] + stream3[2] * b2[i];
}

__global__ __launch_bounds__(256)
void k_init_h(float* __restrict__ h, const float* __restrict__ embed,
              const int* __restrict__ tids, const int* __restrict__ mask) {
  const int m = blockIdx.x;                       // b*CN + n
  const int t = tids[m];
  const float mf = (float)mask[m];
  const float* ep = embed + (size_t)t * CD;
  float* hp = h + (size_t)m * CD;
  for (int d = threadIdx.x; d < CD; d += 256) hp[d] = ep[d] * mf;
}

// ---------------------------------------------------------------------------
// per-step reductions: roles, wsum, rh = sum_n roles*h, mean_ent = rh/wsum
// ---------------------------------------------------------------------------
__global__ __launch_bounds__(256)
void k_reduce(const float* __restrict__ inc, const int* __restrict__ maskI,
              const float* __restrict__ h, float* __restrict__ roles,
              float* __restrict__ rh, float* __restrict__ mean_ent, int s) {
  const int b = blockIdx.x;
  __shared__ float rS[128];
  __shared__ float wsumS;
  const int tid = threadIdx.x;
  if (tid < 128) rS[tid] = 0.f;
  __syncthreads();
  if (tid < CN) {
    float r = inc[((size_t)(b * CN + tid)) * CS + s] * (float)maskI[b * CN + tid];
    rS[tid] = r;
    roles[b * CN + tid] = r;
  }
  __syncthreads();
  if (tid == 0) {
    float w = 0.f;
    for (int n = 0; n < CN; ++n) w += rS[n];
    wsumS = fmaxf(w, 1.0f);
  }
  __syncthreads();
  const float wsum = wsumS;
  for (int d = tid; d < CD; d += 256) {
    float acc = 0.f;
    const float* hp = h + (size_t)b * CN * CD + d;
    for (int n = 0; n < CN; ++n) acc += rS[n] * hp[(size_t)n * CD];
    rh[b * CD + d] = acc;
    mean_ent[b * CD + d] = acc / wsum;
  }
}

// ---------------------------------------------------------------------------
// small WMMA GEMM: M = 16 rows (one WMMA M-tile), Nc = 1024, single block.
// mode 0 (edge):  C = A0@W0^T + A1@W1^T + bias0 + bias1 ; LayerNorm(gamma,beta) -> out
// mode 1 (scene): C = stream3[0] * (A0@W0^T + bias0) -> out
// ---------------------------------------------------------------------------
__global__ __launch_bounds__(256)
void k_small_gemm(const float* __restrict__ A0, int lda0,
                  const float* __restrict__ A1, int lda1,
                  const __bf16* __restrict__ W0, const __bf16* __restrict__ W1,
                  const float* __restrict__ bias0, const float* __restrict__ bias1,
                  const float* __restrict__ gamma, const float* __restrict__ beta,
                  const float* __restrict__ stream3,
                  float* __restrict__ out, int mode) {
  constexpr int LD = 40;
  __shared__ __align__(16) __bf16 As[16 * LD];
  __shared__ float redS[128], redQ[128], meanS[16], ivarS[16];
  const int tid  = threadIdx.x;
  const int w    = tid >> 5;
  const int lane = tid & 31;
  const int ln   = lane & 15, lh = lane >> 4;

  v8f acc[8] = {};

  const int i2 = tid * 2;               // 512 elems, 2 consecutive per thread
  const int ar = i2 >> 5, ac = i2 & 31;

  for (int seg = 0; seg < 2; ++seg) {
    const float*  A = seg ? A1 : A0;
    const __bf16* W = seg ? W1 : W0;
    const int   lda = seg ? lda1 : lda0;
    if (A == nullptr) break;
    for (int k0 = 0; k0 < CD; k0 += 32) {
      __syncthreads();
      {
        const float2 f = *reinterpret_cast<const float2*>(A + (size_t)ar * lda + k0 + ac);
        v2bf pk; pk[0] = (__bf16)f.x; pk[1] = (__bf16)f.y;
        *reinterpret_cast<v2bf*>(As + ar * LD + ac) = pk;
      }
      __syncthreads();
      v16bf a;
      {
        const __bf16* ap = As + ln * LD + 16 * lh;
#pragma unroll
        for (int j = 0; j < 8; ++j) { a[j] = ap[j]; a[8 + j] = ap[8 + j]; }
      }
#pragma unroll
      for (int tn = 0; tn < 8; ++tn) {
        const int n0 = w * 128 + tn * 16;
        const __bf16* bp = W + (size_t)(n0 + ln) * CD + k0 + 16 * lh;
        v16bf bv;
#pragma unroll
        for (int j = 0; j < 16; ++j) bv[j] = bp[j];
        acc[tn] = wmma_bf16(a, bv, acc[tn]);
      }
    }
  }

  if (mode == 1) {
    const float s0 = stream3[0];
#pragma unroll
    for (int tn = 0; tn < 8; ++tn) {
      const int col = w * 128 + tn * 16 + ln;
#pragma unroll
      for (int r = 0; r < 8; ++r) {
        const int row = 8 * lh + r;
        out[(size_t)row * CD + col] = s0 * (acc[tn][r] + bias0[col]);
      }
    }
    return;
  }

  // mode 0: bias + LayerNorm over the 1024 cols of each of 16 rows
  float ps[8], pq[8];
#pragma unroll
  for (int r = 0; r < 8; ++r) { ps[r] = 0.f; pq[r] = 0.f; }
#pragma unroll
  for (int tn = 0; tn < 8; ++tn) {
    const int col = w * 128 + tn * 16 + ln;
    const float bb = bias0[col] + bias1[col];
#pragma unroll
    for (int r = 0; r < 8; ++r) {
      float v = acc[tn][r] + bb;
      acc[tn][r] = v;
      ps[r] += v; pq[r] += v * v;
    }
  }
#pragma unroll
  for (int off = 1; off < 16; off <<= 1) {
#pragma unroll
    for (int r = 0; r < 8; ++r) {
      ps[r] += __shfl_xor(ps[r], off, 32);
      pq[r] += __shfl_xor(pq[r], off, 32);
    }
  }
  if (ln == 0) {
#pragma unroll
    for (int r = 0; r < 8; ++r) {
      redS[w * 16 + 8 * lh + r] = ps[r];
      redQ[w * 16 + 8 * lh + r] = pq[r];
    }
  }
  __syncthreads();
  if (tid < 16) {
    float sm = 0.f, sq = 0.f;
#pragma unroll
    for (int ww = 0; ww < 8; ++ww) { sm += redS[ww * 16 + tid]; sq += redQ[ww * 16 + tid]; }
    const float mu = sm / (float)CD;
    meanS[tid] = mu;
    ivarS[tid] = rsqrtf(sq / (float)CD - mu * mu + 1e-5f);
  }
  __syncthreads();
#pragma unroll
  for (int tn = 0; tn < 8; ++tn) {
    const int col = w * 128 + tn * 16 + ln;
#pragma unroll
    for (int r = 0; r < 8; ++r) {
      const int row = 8 * lh + r;
      out[(size_t)row * CD + col] =
          (acc[tn][r] - meanS[row]) * ivarS[row] * gamma[col] + beta[col];
    }
  }
}

// ---------------------------------------------------------------------------
// generic WMMA GEMM: C[M,NN] = A[M,K](f32) @ W[NN,K]^T(bf16)
// 256 threads = 8 waves; tile 128x64x32; waves 4(M) x 2(N); 2x2 WMMA/wave.
// Double-buffered LDS; B-tiles via async global->LDS DMA (ASYNCcnt),
// A-tiles prefetched to regs, converted fp32->bf16, packed ds_store_b64.
// mode 0: out = acc
// mode 1: out = acc + bias[col]
// mode 2: out = acc + bias[col] + addFull[row,col] + rowVec[row/100, col]
// ---------------------------------------------------------------------------
__global__ __launch_bounds__(256)
void k_gemm(const float* __restrict__ A, int M, int K, int NN,
            const __bf16* __restrict__ W,
            float* __restrict__ Cout,
            const float* __restrict__ bias,
            const float* __restrict__ addFull,
            const float* __restrict__ rowVec,
            int mode) {
  constexpr int BM = 128, BN = 64, BK = 32, LD = 40;
  __shared__ __align__(16) __bf16 As[2 * BM * LD];
  __shared__ __align__(16) __bf16 Bs[2 * BN * LD];
  const int tid  = threadIdx.x;
  const int wave = tid >> 5, lane = tid & 31;
  const int wm = wave & 3, wn = wave >> 2;
  const int ln = lane & 15, lh = lane >> 4;
  const int rowBlk = blockIdx.x * BM;
  const int colBlk = blockIdx.y * BN;

  v8f acc[2][2] = {};

  const int arow0 = tid >> 3;        // 0..31
  const int acol  = (tid & 7) * 4;   // 0..28
  const int brow  = tid >> 2;        // 0..63
  const int bcol  = (tid & 3) * 8;   // 0..24

  const int NT = K / BK;
  float4 areg[4];

  auto loadA = [&](int kt) {
#pragma unroll
    for (int p = 0; p < 4; ++p) {
      const int gr = rowBlk + arow0 + p * 32;
      areg[p] = (gr < M)
                  ? *reinterpret_cast<const float4*>(A + (size_t)gr * K + kt * BK + acol)
                  : make_float4(0.f, 0.f, 0.f, 0.f);
    }
  };
  auto storeA = [&](int buf) {
    __bf16* dst = As + buf * (BM * LD);
#pragma unroll
    for (int p = 0; p < 4; ++p) {
      v4bf pk;
      pk[0] = (__bf16)areg[p].x; pk[1] = (__bf16)areg[p].y;
      pk[2] = (__bf16)areg[p].z; pk[3] = (__bf16)areg[p].w;
      *reinterpret_cast<v4bf*>(dst + (arow0 + p * 32) * LD + acol) = pk;
    }
  };
  auto issueB = [&](int kt, int buf) {
    const unsigned ldst =
        (unsigned)(uintptr_t)(Bs + buf * (BN * LD) + brow * LD + bcol);
    const __bf16* g = W + (size_t)(colBlk + brow) * K + kt * BK + bcol;
    async_load_lds_b128(ldst, g);
  };

  // prologue: tile 0 into buffer 0
  issueB(0, 0);
  loadA(0);
  storeA(0);
  wait_async0();
  __syncthreads();

  for (int kt = 0; kt < NT; ++kt) {
    const int cur = kt & 1, nxt = cur ^ 1;
    if (kt + 1 < NT) {
      issueB(kt + 1, nxt);   // async DMA overlaps with WMMA below
      loadA(kt + 1);
    }
    const __bf16* AsC = As + cur * (BM * LD);
    const __bf16* BsC = Bs + cur * (BN * LD);
    v16bf av[2], bv[2];
#pragma unroll
    for (int t = 0; t < 2; ++t) {
      const __bf16* ap = AsC + (wm * 32 + t * 16 + ln) * LD + 16 * lh;
#pragma unroll
      for (int j = 0; j < 8; ++j) { av[t][j] = ap[j]; av[t][8 + j] = ap[8 + j]; }
      const __bf16* bp = BsC + (wn * 32 + t * 16 + ln) * LD + 16 * lh;
#pragma unroll
      for (int j = 0; j < 16; ++j) bv[t][j] = bp[j];
    }
#pragma unroll
    for (int tm = 0; tm < 2; ++tm)
#pragma unroll
      for (int tn = 0; tn < 2; ++tn)
        acc[tm][tn] = wmma_bf16(av[tm], bv[tn], acc[tm][tn]);
    if (kt + 1 < NT) {
      storeA(nxt);
      wait_async0();
      __syncthreads();
    }
  }

#pragma unroll
  for (int tm = 0; tm < 2; ++tm) {
#pragma unroll
    for (int tn = 0; tn < 2; ++tn) {
      const int col = colBlk + wn * 32 + tn * 16 + ln;
#pragma unroll
      for (int r = 0; r < 8; ++r) {
        const int row = rowBlk + wm * 32 + tm * 16 + 8 * lh + r;
        if (row < M) {
          const size_t idx = (size_t)row * NN + col;
          float v = acc[tm][tn][r];
          if (mode >= 1) v += bias[col];
          if (mode == 2) v += addFull[idx] + rowVec[(size_t)(row / CN) * NN + col];
          Cout[idx] = v;
        }
      }
    }
  }
}

// ---------------------------------------------------------------------------
// arc stream: arc_feat[m,:] = roles[m] * sum_{t<s} inc[m,t]*decay(t) * edges[t,b,:]
// ---------------------------------------------------------------------------
__global__ __launch_bounds__(256)
void k_arcfeat(const float* __restrict__ inc, const float* __restrict__ roles,
               const float* __restrict__ edges, float* __restrict__ arc_feat, int s) {
  const int m = blockIdx.x;        // b*CN + n
  const int b = m / CN;
  const float role = roles[m];
  const int tid = threadIdx.x;
  float acc[4] = {0.f, 0.f, 0.f, 0.f};
  if (role != 0.f && s > 0) {
    const float denom = fmaxf((float)(s - 1), 1.0f);
    const float* incRow = inc + (size_t)m * CS;
    for (int t = 0; t < s; ++t) {
      const float iv = incRow[t];
      if (iv == 0.f) continue;
      const float coef = iv * role * __expf(-1.0f + (float)t / denom);
      const float* ep = edges + ((size_t)t * CB + b) * CD + tid;
#pragma unroll
      for (int j = 0; j < 4; ++j) acc[j] += coef * ep[j * 256];
    }
  }
  float* op = arc_feat + (size_t)m * CD + tid;
#pragma unroll
  for (int j = 0; j < 4; ++j) op[j * 256] = acc[j];
}

// social[m,d] = roles[m]*rh[b,d] - roles[m]^2 * h[m,d]
__global__ __launch_bounds__(256)
void k_social(const float* __restrict__ roles, const float* __restrict__ rh,
              const float* __restrict__ h, float* __restrict__ social) {
  const int m = blockIdx.x;
  const int b = m / CN;
  const float r = roles[m];
  const float r2 = r * r;
  const size_t base = (size_t)m * CD;
#pragma unroll
  for (int j = 0; j < 4; ++j) {
    const int d = threadIdx.x + j * 256;
    social[base + d] = r * rh[(size_t)b * CD + d] - r2 * h[base + d];
  }
}

// GRU cell + masked update (biases already folded into gi/gh)
__global__ __launch_bounds__(256)
void k_gru(const float* __restrict__ gi, const float* __restrict__ gh,
           const float* __restrict__ roles, float* __restrict__ h) {
  const int m = blockIdx.x;
  const float role = roles[m];
  const size_t g0 = (size_t)m * 3 * CD;
  const size_t h0 = (size_t)m * CD;
#pragma unroll
  for (int j = 0; j < 4; ++j) {
    const int d = threadIdx.x + j * 256;
    const float ir = gi[g0 + d],            hr = gh[g0 + d];
    const float iz = gi[g0 + CD + d],       hz = gh[g0 + CD + d];
    const float in_ = gi[g0 + 2 * CD + d],  hn = gh[g0 + 2 * CD + d];
    const float r  = 1.f / (1.f + __expf(-(ir + hr)));
    const float z  = 1.f / (1.f + __expf(-(iz + hz)));
    const float nn = tanhf(in_ + r * hn);
    const float hv = h[h0 + d];
    const float hnew = (1.f - z) * nn + z * hv;
    h[h0 + d] = (role > 0.f) ? hnew : hv;
  }
}

// ---------------------------------------------------------------------------
// final stage
// ---------------------------------------------------------------------------
__device__ __forceinline__ void block_reduce2(float& sm, float& sq, float* red) {
  const int tid = threadIdx.x;
  red[tid] = sm; red[256 + tid] = sq;
  __syncthreads();
  for (int off = 128; off > 0; off >>= 1) {
    if (tid < off) { red[tid] += red[tid + off]; red[256 + tid] += red[256 + tid + off]; }
    __syncthreads();
  }
  sm = red[0]; sq = red[256];
  __syncthreads();
}

__global__ __launch_bounds__(256)
void k_resln(const float* __restrict__ edges, const float* __restrict__ scene,
             const float* __restrict__ gamma, const float* __restrict__ beta,
             float* __restrict__ H1) {
  __shared__ float red[512];
  const int row = blockIdx.x;              // b*CS + s
  const int b = row / CS, s = row % CS;
  const float* e  = edges + ((size_t)s * CB + b) * CD;
  const float* sc = scene + (size_t)row * CD;
  const int tid = threadIdx.x;
  float vals[4];
  float sm = 0.f, sq = 0.f;
#pragma unroll
  for (int j = 0; j < 4; ++j) {
    const int d = tid + j * 256;
    const float v = e[d] + sc[d];
    vals[j] = v; sm += v; sq += v * v;
  }
  block_reduce2(sm, sq, red);
  const float mu = sm / (float)CD;
  const float iv = rsqrtf(sq / (float)CD - mu * mu + 1e-5f);
#pragma unroll
  for (int j = 0; j < 4; ++j) {
    const int d = tid + j * 256;
    H1[(size_t)row * CD + d] = (vals[j] - mu) * iv * gamma[d] + beta[d];
  }
}

__global__ __launch_bounds__(256)
void k_outln(const float* __restrict__ G, const float* __restrict__ gamma,
             const float* __restrict__ beta, float* __restrict__ Hout) {
  __shared__ float red[512];
  const int row = blockIdx.x;
  const int tid = threadIdx.x;
  float vals[4];
  float sm = 0.f, sq = 0.f;
#pragma unroll
  for (int j = 0; j < 4; ++j) {
    const int d = tid + j * 256;
    const float x = G[(size_t)row * CD + d];
    const float v = 0.5f * x * (1.f + erff(x * 0.70710678118654752440f));  // exact GELU
    vals[j] = v; sm += v; sq += v * v;
  }
  block_reduce2(sm, sq, red);
  const float mu = sm / (float)CD;
  const float iv = rsqrtf(sq / (float)CD - mu * mu + 1e-5f);
#pragma unroll
  for (int j = 0; j < 4; ++j) {
    const int d = tid + j * 256;
    Hout[(size_t)row * CD + d] = (vals[j] - mu) * iv * gamma[d] + beta[d];
  }
}

// ---------------------------------------------------------------------------
// host driver
// ---------------------------------------------------------------------------
extern "C" void kernel_launch(void* const* d_in, const int* in_sizes, int n_in,
                              void* d_out, int out_size, void* d_ws, size_t ws_size,
                              hipStream_t stream) {
  (void)in_sizes; (void)n_in; (void)out_size; (void)ws_size;
  const float* scene_reps = (const float*)d_in[0];
  const float* inc        = (const float*)d_in[1];
  // d_in[2] edge_type_ids: unused (matches reference)
  const int*   ent_type   = (const int*)d_in[3];
  const int*   ent_mask   = (const int*)d_in[4];
  const float* embed      = (const float*)d_in[5];
  const float* W_node = (const float*)d_in[6];  const float* b_node = (const float*)d_in[7];
  const float* W_text = (const float*)d_in[8];  const float* b_text = (const float*)d_in[9];
  const float* g_edge = (const float*)d_in[10]; const float* b_edge = (const float*)d_in[11];
  const float* stream_w = (const float*)d_in[12];
  const float* W_scene = (const float*)d_in[13]; const float* b_scene = (const float*)d_in[14];
  const float* W_arc = (const float*)d_in[15];  const float* b_arc = (const float*)d_in[16];
  const float* W_int = (const float*)d_in[17];  const float* b_int = (const float*)d_in[18];
  const float* W_ih = (const float*)d_in[19];   const float* b_ih = (const float*)d_in[20];
  const float* W_hh = (const float*)d_in[21];   const float* b_hh = (const float*)d_in[22];
  const float* g_res = (const float*)d_in[23];  const float* b_res = (const float*)d_in[24];
  const float* W_out = (const float*)d_in[25];  const float* b_out = (const float*)d_in[26];
  const float* g_outln = (const float*)d_in[27]; const float* b_outln = (const float*)d_in[28];

  float* Hout = (float*)d_out;                          // [CB,CS,CD]
  float* h    = Hout + (size_t)CB * CS * CD;            // [CB,CN,CD] == h_final region

  // workspace carve-up (256B aligned chunks)
  size_t off = 0;
  auto alloc = [&](size_t bytes) -> void* {
    void* p = (char*)d_ws + off;
    off += (bytes + 255) & ~(size_t)255;
    return p;
  };
  const size_t DD = (size_t)CD * CD;
  __bf16* Wn_b   = (__bf16*)alloc(DD * 2);
  __bf16* Wt_b   = (__bf16*)alloc(DD * 2);
  __bf16* Ws_b   = (__bf16*)alloc(DD * 2);
  __bf16* Wa_b   = (__bf16*)alloc(DD * 2);       // pre-scaled by softmax[1]
  __bf16* Wi_b   = (__bf16*)alloc(DD * 2);       // pre-scaled by softmax[2]
  __bf16* Wout_b = (__bf16*)alloc(DD * 2);
  __bf16* Wih_b  = (__bf16*)alloc(3 * DD * 2);
  __bf16* Whh_b  = (__bf16*)alloc(3 * DD * 2);
  float* stream3  = (float*)alloc(16);
  float* bArcInt  = (float*)alloc(CD * 4);
  float* roles    = (float*)alloc((size_t)CB * CN * 4);
  float* rh       = (float*)alloc((size_t)CB * CD * 4);
  float* mean_ent = (float*)alloc((size_t)CB * CD * 4);
  float* msg_scene= (float*)alloc((size_t)CB * CD * 4);
  float* edges    = (float*)alloc((size_t)CS * CB * CD * 4);     // [S][B][D]
  float* arc_feat = (float*)alloc((size_t)CB * CN * CD * 4);
  float* social   = (float*)alloc((size_t)CB * CN * CD * 4);
  float* msgA     = (float*)alloc((size_t)CB * CN * CD * 4);
  float* msg      = (float*)alloc((size_t)CB * CN * CD * 4);
  float* gi       = (float*)alloc((size_t)CB * CN * 3 * CD * 4);
  float* gh       = (float*)alloc((size_t)CB * CN * 3 * CD * 4);
  float* H1       = (float*)alloc((size_t)CB * CS * CD * 4);
  float* G        = (float*)alloc((size_t)CB * CS * CD * 4);

  // ---- one-time prep ----
  k_stream<<<1, 32, 0, stream>>>(stream_w, stream3);
  auto cv = [&](__bf16* dst, const float* src, size_t n, int sidx) {
    k_convert<<<(unsigned)((n + 255) / 256), 256, 0, stream>>>(dst, src, (int)n, stream3, sidx);
  };
  cv(Wn_b, W_node, DD, -1);
  cv(Wt_b, W_text, DD, -1);
  cv(Ws_b, W_scene, DD, -1);
  cv(Wa_b, W_arc, DD, 1);
  cv(Wi_b, W_int, DD, 2);
  cv(Wout_b, W_out, DD, -1);
  cv(Wih_b, W_ih, 3 * DD, -1);
  cv(Whh_b, W_hh, 3 * DD, -1);
  k_bias2<<<CD / 256, 256, 0, stream>>>(bArcInt, b_arc, b_int, stream3, CD);
  k_init_h<<<CB * CN, 256, 0, stream>>>(h, embed, ent_type, ent_mask);

  const int M = CB * CN;                        // 1600
  const dim3 gD((M + 127) / 128, CD / 64);      // 13 x 16
  const dim3 g3D((M + 127) / 128, 3 * CD / 64); // 13 x 48

  // ---- sequential scan over scene steps ----
  for (int s = 0; s < CS; ++s) {
    k_reduce<<<CB, 256, 0, stream>>>(inc, ent_mask, h, roles, rh, mean_ent, s);
    // e_s = LN(mean_ent@Wn^T + text@Wt^T + b) -> edges[s]
    k_small_gemm<<<1, 256, 0, stream>>>(mean_ent, CD,
                                        scene_reps + (size_t)s * CD, CS * CD,
                                        Wn_b, Wt_b, b_node, b_text,
                                        g_edge, b_edge, nullptr,
                                        edges + (size_t)s * CB * CD, 0);
    // msg_scene = s0*(e_s@Ws^T + b_scene)
    k_small_gemm<<<1, 256, 0, stream>>>(edges + (size_t)s * CB * CD, CD,
                                        nullptr, 0,
                                        Ws_b, nullptr, b_scene, nullptr,
                                        nullptr, nullptr, stream3,
                                        msg_scene, 1);
    k_arcfeat<<<CB * CN, 256, 0, stream>>>(inc, roles, edges, arc_feat, s);
    k_social<<<CB * CN, 256, 0, stream>>>(roles, rh, h, social);
    // msgA = arc_feat @ (s1*W_arc)^T
    k_gemm<<<gD, 256, 0, stream>>>(arc_feat, M, CD, CD, Wa_b, msgA,
                                   nullptr, nullptr, nullptr, 0);
    // msg = social @ (s2*W_int)^T + (s1*b_arc + s2*b_int) + msgA + msg_scene[b]
    k_gemm<<<gD, 256, 0, stream>>>(social, M, CD, CD, Wi_b, msg,
                                   bArcInt, msgA, msg_scene, 2);
    // gi = msg @ W_ih^T + b_ih ; gh = h @ W_hh^T + b_hh
    k_gemm<<<g3D, 256, 0, stream>>>(msg, M, CD, 3 * CD, Wih_b, gi,
                                    b_ih, nullptr, nullptr, 1);
    k_gemm<<<g3D, 256, 0, stream>>>(h, M, CD, 3 * CD, Whh_b, gh,
                                    b_hh, nullptr, nullptr, 1);
    k_gru<<<CB * CN, 256, 0, stream>>>(gi, gh, roles, h);
  }

  // ---- output head ----
  k_resln<<<CB * CS, 256, 0, stream>>>(edges, scene_reps, g_res, b_res, H1);
  k_gemm<<<dim3((CB * CS + 127) / 128, CD / 64), 256, 0, stream>>>(
      H1, CB * CS, CD, CD, Wout_b, G, b_out, nullptr, nullptr, 1);
  k_outln<<<CB * CS, 256, 0, stream>>>(G, g_outln, b_outln, Hout);
}
